// MDLRNN_48954037239828
// MI455X (gfx1250) — matrix operand order repaired and checked by reading, not compiled, not run on hardware
//
#include <hip/hip_runtime.h>
#include <math.h>

// ---------------------------------------------------------------------------
// MDL-RNN for MI455X (gfx1250, wave32, WMMA bf16 16x16x32).
// Phase 1: hoisted input GEMMs (parallel).  Phase 2: persistent 16-WG scan.
// ---------------------------------------------------------------------------

typedef __bf16  v16bf __attribute__((ext_vector_type(16)));
typedef __bf16  v8bf  __attribute__((ext_vector_type(8)));
typedef float   v8f   __attribute__((ext_vector_type(8)));

#define B_SZ   64
#define T_SZ   512
#define DIN    512
#define H_SZ   1024
#define DOUT   512
#define M_SZ   256
#define NWG    16

__device__ __forceinline__ v8f wmma_bf16(v16bf a, v16bf b, v8f c) {
    // D = A(16x32 bf16) x B(32x16 bf16) + C(16x16 f32)
    return __builtin_amdgcn_wmma_f32_16x16x32_bf16(
        /*neg_a=*/false, a, /*neg_b=*/false, b,
        /*c_mod=*/(short)0, c, /*reuse_a=*/false, /*reuse_b=*/false);
}

// A fragment 16x32 (MxK) from row-major [rows, row_stride] source.
// Layout: lanes 0-15 (M=lane): elems 0-15 -> K 0..15
//         lanes16-31 (M=lane-16): elems 0-15 -> K 16..31
__device__ __forceinline__ v16bf load_a_frag(const __bf16* base, int row_stride, int lane) {
    const int m  = lane & 15;
    const int hi = lane >> 4;
    const __bf16* p = base + (size_t)m * row_stride;
    v8bf lo = *reinterpret_cast<const v8bf*>(p + (hi ? 16 : 0));
    v8bf up = *reinterpret_cast<const v8bf*>(p + (hi ? 24 : 8));
    return __builtin_shufflevector(lo, up, 0,1,2,3,4,5,6,7,8,9,10,11,12,13,14,15);
}

// B fragment 32x16 (KxN), B[k][n] = W[n][k], W row-major [N, row_stride].
// Layout (same k order as A): lanes 0-15 (N=lane) hold K 0..15; lanes 16-31 hold K 16..31.
__device__ __forceinline__ v16bf load_b_frag(const __bf16* base, int row_stride, int lane) {
    const int n  = lane & 15;
    const int hi = lane >> 4;
    return *reinterpret_cast<const v16bf*>(base + (size_t)n * row_stride + (hi ? 16 : 0));
}

// ---------------------------------------------------------------------------
// Prep kernels
// ---------------------------------------------------------------------------
__global__ void f2bf_kernel(const float* __restrict__ src,
                            unsigned short* __restrict__ dst, int n) {
    int i = blockIdx.x * blockDim.x + threadIdx.x;
    if (i < n) {
        __bf16 h = (__bf16)src[i];
        dst[i] = *reinterpret_cast<unsigned short*>(&h);
    }
}

__global__ void prep_misc_kernel(const float* __restrict__ b01, const float* __restrict__ bm1,
                                 const float* __restrict__ b02, const float* __restrict__ b12,
                                 const float* __restrict__ bm2,
                                 float* __restrict__ b1c, float* __restrict__ b2c,
                                 unsigned* __restrict__ counters) {
    int i = threadIdx.x;                    // one block of 1024
    b1c[i] = b01[i] + bm1[i];
    if (i < DOUT) b2c[i] = b02[i] + b12[i] + bm2[i];
    counters[i] = 0u;                       // 2 barriers x 512 steps
}

// ---------------------------------------------------------------------------
// Phase 1: Y[row(t,b), n] = Xbf[r,:] @ Wbf[n,:] + bias[n], stored as [t*64+b, N]
// 64x64 tile per 256-thread WG (8 waves, 2 row-tiles each, B-frag reuse).
// ---------------------------------------------------------------------------
template <bool OUT_F32>
__global__ void __launch_bounds__(256)
gemm_xw_kernel(const unsigned short* __restrict__ Ap,   // Xbf [B*T, K]
               const unsigned short* __restrict__ Wp,   // Wbf [N, K]
               const float* __restrict__ bias,          // [N]
               unsigned short* __restrict__ outb,       // bf16 out [T*B, N]
               float* __restrict__ outf,                // f32  out [T*B, N]
               int K, int N) {
    const __bf16* A = reinterpret_cast<const __bf16*>(Ap);
    const __bf16* W = reinterpret_cast<const __bf16*>(Wp);
    const int lane = threadIdx.x & 31;
    const int wave = threadIdx.x >> 5;          // 0..7
    const int rowbase = blockIdx.x * 64;
    const int colbase = blockIdx.y * 64;
    const int ct  = wave & 3;
    const int rt0 = (wave >> 2) * 2;            // {0,2}
    const int colb = colbase + ct * 16;
    const int n = colb + (lane & 15);

    const float bv = bias[n];
    v8f c0 = { bv, bv, bv, bv, bv, bv, bv, bv };
    v8f c1 = c0;

    for (int k = 0; k < K; k += 32) {
        v16bf bf = load_b_frag(W + (size_t)colb * K + k, K, lane);
        v16bf a0 = load_a_frag(A + (size_t)(rowbase + rt0 * 16) * K + k, K, lane);
        v16bf a1 = load_a_frag(A + (size_t)(rowbase + rt0 * 16 + 16) * K + k, K, lane);
        c0 = wmma_bf16(a0, bf, c0);
        c1 = wmma_bf16(a1, bf, c1);
    }
    // C/D layout: VGPR r, lanes 0-15 -> M=r, lanes 16-31 -> M=r+8; N = lane&15.
    #pragma unroll
    for (int r = 0; r < 8; ++r) {
        const int M  = r + ((lane >> 4) << 3);
        const int R0 = rowbase + rt0 * 16 + M;        // r = b*T + t
        const int R1 = R0 + 16;
        const int b0 = R0 >> 9, t0 = R0 & 511;
        const int b1 = R1 >> 9, t1 = R1 & 511;
        const size_t o0 = ((size_t)t0 * B_SZ + b0) * N + n;
        const size_t o1 = ((size_t)t1 * B_SZ + b1) * N + n;
        if (OUT_F32) { outf[o0] = c0[r]; outf[o1] = c1[r]; }
        else {
            __bf16 h0 = (__bf16)c0[r], h1 = (__bf16)c1[r];
            outb[o0] = *reinterpret_cast<unsigned short*>(&h0);
            outb[o1] = *reinterpret_cast<unsigned short*>(&h1);
        }
    }
}

// ---------------------------------------------------------------------------
// Device-scope barrier (one counter per barrier instance; pre-zeroed per call)
// ---------------------------------------------------------------------------
__device__ __forceinline__ void grid_sync(unsigned* counters, int idx) {
    __threadfence();
    __syncthreads();
    if (threadIdx.x == 0) {
        __hip_atomic_fetch_add(&counters[idx], 1u, __ATOMIC_RELEASE, __HIP_MEMORY_SCOPE_AGENT);
        while (__hip_atomic_load(&counters[idx], __ATOMIC_ACQUIRE, __HIP_MEMORY_SCOPE_AGENT)
               < (unsigned)NWG) {
            __builtin_amdgcn_s_sleep(2);
        }
    }
    __syncthreads();
    __threadfence();
}

// ---------------------------------------------------------------------------
// Phase 2: persistent scan. 16 WGs x 256 threads.
//  Stage A: WG w -> v1 cols [w*64, w*64+64), K=256 (mem in LDS)
//  Stage B: WG w -> v2 cols [w*32, w*32+32), K=1024 (v1) + K=256 (mem in LDS)
// ---------------------------------------------------------------------------
__global__ void __launch_bounds__(256)
rnn_scan_kernel(const unsigned short* __restrict__ Y1p,  // bf16 [T*B, 1024]
                const float* __restrict__ Y2,            // f32  [T*B, 512]
                const unsigned short* __restrict__ Wm1p, // bf16 [1024, 256]
                const unsigned short* __restrict__ W12p, // bf16 [512, 1024]
                const unsigned short* __restrict__ Wm2p, // bf16 [512, 256]
                unsigned short* __restrict__ v1p,        // bf16 [64, 1024]
                unsigned short* __restrict__ mem0p,      // bf16 [64, 256]
                unsigned short* __restrict__ mem1p,      // bf16 [64, 256]
                unsigned* __restrict__ counters,
                float* __restrict__ out) {               // f32 [B, T, 512]
    const __bf16* Y1  = reinterpret_cast<const __bf16*>(Y1p);
    const __bf16* Wm1 = reinterpret_cast<const __bf16*>(Wm1p);
    const __bf16* W12 = reinterpret_cast<const __bf16*>(W12p);
    const __bf16* Wm2 = reinterpret_cast<const __bf16*>(Wm2p);
    __bf16* v1buf = reinterpret_cast<__bf16*>(v1p);
    __bf16* mem0  = reinterpret_cast<__bf16*>(mem0p);
    __bf16* mem1  = reinterpret_cast<__bf16*>(mem1p);

    __shared__ __bf16 s_mem[B_SZ * M_SZ];      // 32 KB, current-step memory

    const int tid  = threadIdx.x;
    const int lane = tid & 31;
    const int wave = tid >> 5;                 // 0..7
    const int w    = blockIdx.x;               // 0..15

    for (int t = 0; t < T_SZ; ++t) {
        const __bf16* memA = (t & 1) ? mem1 : mem0;
        __bf16*       memB = (t & 1) ? mem0 : mem1;

        // Stage current memory into LDS (16384 bf16 = 2048 x v8bf).
        {
            const v8bf* src = reinterpret_cast<const v8bf*>(memA);
            v8bf*       dst = reinterpret_cast<v8bf*>(s_mem);
            #pragma unroll
            for (int i = tid; i < (B_SZ * M_SZ) / 8; i += 256) dst[i] = src[i];
        }
        __syncthreads();

        // ---- Stage A: v1 slice ----
        {
            const int ct   = wave & 3;
            const int rt0  = (wave >> 2) * 2;
            const int colb = w * 64 + ct * 16;
            const int n    = colb + (lane & 15);
            const __bf16* y1t = Y1 + (size_t)t * B_SZ * H_SZ;

            v8f c0, c1;
            #pragma unroll
            for (int r = 0; r < 8; ++r) {
                const int M = r + ((lane >> 4) << 3);
                c0[r] = (float)y1t[(size_t)(rt0 * 16 + M) * H_SZ + n];
                c1[r] = (float)y1t[(size_t)(rt0 * 16 + 16 + M) * H_SZ + n];
            }
            #pragma unroll
            for (int k = 0; k < M_SZ; k += 32) {
                v16bf bf = load_b_frag(Wm1 + (size_t)colb * M_SZ + k, M_SZ, lane);
                v16bf a0 = load_a_frag(s_mem + (rt0 * 16) * M_SZ + k, M_SZ, lane);
                v16bf a1 = load_a_frag(s_mem + (rt0 * 16 + 16) * M_SZ + k, M_SZ, lane);
                c0 = wmma_bf16(a0, bf, c0);
                c1 = wmma_bf16(a1, bf, c1);
            }
            const bool use_tanh = (colb < 512);   // TANH_N, tile-uniform
            const bool to_mem   = (colb < 128);   // M1, tile-uniform
            #pragma unroll
            for (int r = 0; r < 8; ++r) {
                const int M  = r + ((lane >> 4) << 3);
                const int b0 = rt0 * 16 + M;
                const int b1 = b0 + 16;
                const float x0 = use_tanh ? tanhf(c0[r]) : fmaxf(c0[r], 0.0f);
                const float x1 = use_tanh ? tanhf(c1[r]) : fmaxf(c1[r], 0.0f);
                const __bf16 h0 = (__bf16)x0, h1 = (__bf16)x1;
                v1buf[(size_t)b0 * H_SZ + n] = h0;
                v1buf[(size_t)b1 * H_SZ + n] = h1;
                if (to_mem) {                      // mem' part 1: v1[:, :128]
                    memB[b0 * M_SZ + n] = h0;
                    memB[b1 * M_SZ + n] = h1;
                }
            }
        }
        grid_sync(counters, 2 * t);

        // ---- Stage B: v2 slice ----
        {
            const int ct   = wave >> 2;            // 0..1
            const int rt   = wave & 3;             // 0..3
            const int colb = w * 32 + ct * 16;
            const int n    = colb + (lane & 15);
            const float* y2t = Y2 + (size_t)t * B_SZ * DOUT;

            v8f c;
            #pragma unroll
            for (int r = 0; r < 8; ++r) {
                const int M = r + ((lane >> 4) << 3);
                c[r] = y2t[(size_t)(rt * 16 + M) * DOUT + n];
            }
            for (int k = 0; k < H_SZ; k += 32) {   // v1 @ W12^T
                v16bf bf = load_b_frag(W12 + (size_t)colb * H_SZ + k, H_SZ, lane);
                v16bf a  = load_a_frag(v1buf + (size_t)(rt * 16) * H_SZ + k, H_SZ, lane);
                c = wmma_bf16(a, bf, c);
            }
            #pragma unroll
            for (int k = 0; k < M_SZ; k += 32) {   // mem @ Wm2^T
                v16bf bf = load_b_frag(Wm2 + (size_t)colb * M_SZ + k, M_SZ, lane);
                v16bf a  = load_a_frag(s_mem + (rt * 16) * M_SZ + k, M_SZ, lane);
                c = wmma_bf16(a, bf, c);
            }
            const bool sig = (colb < 128);         // SIG_N, tile-uniform
            #pragma unroll
            for (int r = 0; r < 8; ++r) {
                const int M = r + ((lane >> 4) << 3);
                const int b = rt * 16 + M;
                float x = c[r];
                if (sig) x = 1.0f / (1.0f + __expf(-x));
                out[((size_t)b * T_SZ + t) * DOUT + n] = x;
                if (sig) {                         // mem' part 2: v2[:, :128]
                    __bf16 h = (__bf16)x;
                    memB[b * M_SZ + 128 + n] = h;
                }
            }
        }
        grid_sync(counters, 2 * t + 1);
    }
}

// ---------------------------------------------------------------------------
// Host launcher
// ---------------------------------------------------------------------------
extern "C" void kernel_launch(void* const* d_in, const int* in_sizes, int n_in,
                              void* d_out, int out_size, void* d_ws, size_t ws_size,
                              hipStream_t stream) {
    (void)in_sizes; (void)n_in; (void)out_size; (void)ws_size;
    const float* X    = (const float*)d_in[0];
    const float* memI = (const float*)d_in[1];
    const float* W01  = (const float*)d_in[2];
    const float* b01  = (const float*)d_in[3];
    const float* W02  = (const float*)d_in[4];
    const float* b02  = (const float*)d_in[5];
    const float* W12  = (const float*)d_in[6];
    const float* b12  = (const float*)d_in[7];
    const float* Wm1  = (const float*)d_in[8];
    const float* bm1  = (const float*)d_in[9];
    const float* Wm2  = (const float*)d_in[10];
    const float* bm2  = (const float*)d_in[11];
    float* out = (float*)d_out;

    char* p = (char*)d_ws;
    auto alloc = [&](size_t bytes) -> char* {
        char* r = p; p += (bytes + 255) & ~(size_t)255; return r;
    };
    unsigned short* Xbf   = (unsigned short*)alloc((size_t)B_SZ * T_SZ * DIN * 2);
    unsigned short* W01bf = (unsigned short*)alloc((size_t)H_SZ * DIN * 2);
    unsigned short* W02bf = (unsigned short*)alloc((size_t)DOUT * DIN * 2);
    unsigned short* W12bf = (unsigned short*)alloc((size_t)DOUT * H_SZ * 2);
    unsigned short* Wm1bf = (unsigned short*)alloc((size_t)H_SZ * M_SZ * 2);
    unsigned short* Wm2bf = (unsigned short*)alloc((size_t)DOUT * M_SZ * 2);
    float*          b1c   = (float*)alloc(H_SZ * 4);
    float*          b2c   = (float*)alloc(DOUT * 4);
    unsigned short* Y1bf  = (unsigned short*)alloc((size_t)B_SZ * T_SZ * H_SZ * 2);
    float*          Y2f   = (float*)alloc((size_t)B_SZ * T_SZ * DOUT * 4);
    unsigned short* v1buf = (unsigned short*)alloc((size_t)B_SZ * H_SZ * 2);
    unsigned short* mem0  = (unsigned short*)alloc((size_t)B_SZ * M_SZ * 2);
    unsigned short* mem1  = (unsigned short*)alloc((size_t)B_SZ * M_SZ * 2);
    unsigned*       cnts  = (unsigned*)alloc(2 * T_SZ * 4);

    auto cvt = [&](const float* s, unsigned short* d, int n) {
        f2bf_kernel<<<(n + 255) / 256, 256, 0, stream>>>(s, d, n);
    };
    cvt(X,    Xbf,   B_SZ * T_SZ * DIN);
    cvt(W01,  W01bf, H_SZ * DIN);
    cvt(W02,  W02bf, DOUT * DIN);
    cvt(W12,  W12bf, DOUT * H_SZ);
    cvt(Wm1,  Wm1bf, H_SZ * M_SZ);
    cvt(Wm2,  Wm2bf, DOUT * M_SZ);
    cvt(memI, mem0,  B_SZ * M_SZ);

    prep_misc_kernel<<<1, 1024, 0, stream>>>(b01, bm1, b02, b12, bm2, b1c, b2c, cnts);

    // Phase 1: hoisted input GEMMs.
    gemm_xw_kernel<false><<<dim3((B_SZ * T_SZ) / 64, H_SZ / 64), 256, 0, stream>>>(
        Xbf, W01bf, b1c, Y1bf, nullptr, DIN, H_SZ);
    gemm_xw_kernel<true><<<dim3((B_SZ * T_SZ) / 64, DOUT / 64), 256, 0, stream>>>(
        Xbf, W02bf, b2c, nullptr, Y2f, DIN, DOUT);

    // Phase 2: persistent recurrent scan.
    rnn_scan_kernel<<<NWG, 256, 0, stream>>>(
        Y1bf, Y2f, Wm1bf, W12bf, Wm2bf, v1buf, mem0, mem1, cnts, out);
}
